// MultiHeadAttention_87308095193127
// MI455X (gfx1250) — matrix operand order repaired and checked by reading, not compiled, not run on hardware
//
#include <hip/hip_runtime.h>

// ---------------- CDNA5 WMMA types/helpers ----------------
typedef __attribute__((ext_vector_type(16))) __bf16       v16bf;
typedef __attribute__((ext_vector_type(8)))  float        v8f;
typedef __attribute__((ext_vector_type(8)))  unsigned int v8u;

static __device__ __forceinline__ unsigned short f32_to_bf16(float f) {
  unsigned int u = __float_as_uint(f);
  u += 0x7FFFu + ((u >> 16) & 1u);           // round-to-nearest-even
  return (unsigned short)(u >> 16);
}
static __device__ __forceinline__ unsigned int pack2_bf16(float a, float b) {
  unsigned int ua = __float_as_uint(a); ua += 0x7FFFu + ((ua >> 16) & 1u);
  unsigned int ub = __float_as_uint(b); ub += 0x7FFFu + ((ub >> 16) & 1u);
  return (ua >> 16) | ((ub >> 16) << 16);
}
static __device__ __forceinline__ v16bf frag_cast(const unsigned int* w) {
  v8u t;
#pragma unroll
  for (int j = 0; j < 8; ++j) t[j] = w[j];
  return __builtin_bit_cast(v16bf, t);
}
static __device__ __forceinline__ v8f wmma_bf16(const unsigned int* a,
                                                const unsigned int* b, v8f c) {
  return __builtin_amdgcn_wmma_f32_16x16x32_bf16(
      false, frag_cast(a), false, frag_cast(b), (short)0, c, false, false);
}

// LDS byte offset of a generic __shared__ pointer: per flat-aperture rules the
// low 32 bits of a generic LDS address are the wave-relative LDS offset.
static __device__ __forceinline__ unsigned int lds_off(const void* p) {
  return (unsigned int)(size_t)p;
}
// Async DMA one dword global -> LDS (tracked by ASYNCcnt).
static __device__ __forceinline__ void async_g2l_b32(unsigned int ldsOff,
                                                     const void* gaddr) {
  asm volatile("global_load_async_to_lds_b32 %0, %1, off"
               :: "v"(ldsOff), "v"((unsigned long long)(size_t)gaddr)
               : "memory");
}
static __device__ __forceinline__ void wait_async0() {
  asm volatile("s_wait_asynccnt 0x0" ::: "memory");
}

// A-fragment (16x32 bf16, MxK): per lane one contiguous 16-element K run
// (kbase..kbase+15 -> words 0..7)
static __device__ __forceinline__ void load_fragA(const float* p, unsigned int* d) {
  const float4 f0 = *reinterpret_cast<const float4*>(p);
  const float4 f1 = *reinterpret_cast<const float4*>(p + 4);
  const float4 f2 = *reinterpret_cast<const float4*>(p + 8);
  const float4 f3 = *reinterpret_cast<const float4*>(p + 12);
  d[0] = pack2_bf16(f0.x, f0.y); d[1] = pack2_bf16(f0.z, f0.w);
  d[2] = pack2_bf16(f1.x, f1.y); d[3] = pack2_bf16(f1.z, f1.w);
  d[4] = pack2_bf16(f2.x, f2.y); d[5] = pack2_bf16(f2.z, f2.w);
  d[6] = pack2_bf16(f3.x, f3.y); d[7] = pack2_bf16(f3.z, f3.w);
}
static __device__ __forceinline__ void load_fragA(const unsigned short* p, unsigned int* d) {
  const uint4 u0 = *reinterpret_cast<const uint4*>(p);
  const uint4 u1 = *reinterpret_cast<const uint4*>(p + 8);
  d[0] = u0.x; d[1] = u0.y; d[2] = u0.z; d[3] = u0.w;
  d[4] = u1.x; d[5] = u1.y; d[6] = u1.z; d[7] = u1.w;
}
// 16 consecutive bf16 from memory -> 8 packed words (B fragment half-row)
static __device__ __forceinline__ void load_b16(const unsigned short* p, unsigned int* d) {
  const uint4 u0 = *reinterpret_cast<const uint4*>(p);
  const uint4 u1 = *reinterpret_cast<const uint4*>(p + 8);
  d[0] = u0.x; d[1] = u0.y; d[2] = u0.z; d[3] = u0.w;
  d[4] = u1.x; d[5] = u1.y; d[6] = u1.z; d[7] = u1.w;
}
// 16 consecutive f32 -> 8 packed bf16 words
static __device__ __forceinline__ void pack_b16_from_f32(const float* p, unsigned int* d) {
  const float4 f0 = *reinterpret_cast<const float4*>(p);
  const float4 f1 = *reinterpret_cast<const float4*>(p + 4);
  const float4 f2 = *reinterpret_cast<const float4*>(p + 8);
  const float4 f3 = *reinterpret_cast<const float4*>(p + 12);
  d[0] = pack2_bf16(f0.x, f0.y); d[1] = pack2_bf16(f0.z, f0.w);
  d[2] = pack2_bf16(f1.x, f1.y); d[3] = pack2_bf16(f1.z, f1.w);
  d[4] = pack2_bf16(f2.x, f2.y); d[5] = pack2_bf16(f2.z, f2.w);
  d[6] = pack2_bf16(f3.x, f3.y); d[7] = pack2_bf16(f3.z, f3.w);
}

static __device__ __forceinline__ void store_out(float* p, float v) { *p = v; }
static __device__ __forceinline__ void store_out(unsigned short* p, float v) { *p = f32_to_bf16(v); }

// ---------------- Generic GEMM: C[M,N] = X[M,K] * W[K,N] ----------------
// Block: 256 thr = 8 waves, tile M=64 x N=64 (waves 4x2, each wave 16x32,
// 2 WMMAs per K-step sharing one A fragment).
// W tile [32k x 64n] f32 staged TRANSPOSED into LDS via per-lane async DMA
// scatter (global_load_async_to_lds_b32): lBf[n*36 + k].
template <typename InT, typename OutT>
__global__ void gemm_bf16_wmma(const InT* __restrict__ X, const float* __restrict__ W,
                               OutT* __restrict__ C, int M, int N, int Kd) {
  __shared__ float lBf[64 * 36];
  const int tid = threadIdx.x;
  const int w = tid >> 5, lane = tid & 31;
  const int wave_m = w >> 1, wave_n = w & 1;
  const int laneLo = lane & 15, laneHi = lane >> 4;
  const int nTilesN = N / 64;
  const int bm = blockIdx.x / nTilesN, bn = blockIdx.x % nTilesN;

  const int rowA = bm * 64 + wave_m * 16 + laneLo;
  const InT* arow = X + (size_t)rowA * Kd;
  v8f acc0 = {}, acc1 = {};

  const int nn  = tid & 63;   // staging: col within tile
  const int kp0 = tid >> 6;   // staging: k row (0..3), 8 passes

  for (int kk = 0; kk < Kd; kk += 32) {
    // async DMA scatter-transpose of the W tile into LDS
#pragma unroll
    for (int p = 0; p < 8; ++p) {
      const int k = kp0 + p * 4;
      async_g2l_b32(lds_off(&lBf[nn * 36 + k]),
                    &W[(size_t)(kk + k) * N + bn * 64 + nn]);
    }

    if (kk + 32 < Kd) __builtin_prefetch(arow + kk + 32 + laneHi * 8, 0, 0);

    unsigned int afr[8];
    load_fragA(arow + kk + laneHi * 16, afr);   // overlaps with async DMA

    wait_async0();
    __syncthreads();

    unsigned int bfr0[8], bfr1[8];
    pack_b16_from_f32(&lBf[(wave_n * 32 + laneLo) * 36 + laneHi * 16], bfr0);
    pack_b16_from_f32(&lBf[(wave_n * 32 + 16 + laneLo) * 36 + laneHi * 16], bfr1);

    acc0 = wmma_bf16(afr, bfr0, acc0);
    acc1 = wmma_bf16(afr, bfr1, acc1);
    __syncthreads();
  }

  const int col0 = bn * 64 + wave_n * 32 + laneLo;
#pragma unroll
  for (int r = 0; r < 8; ++r) {
    int row = bm * 64 + wave_m * 16 + r + laneHi * 8;
    store_out(C + (size_t)row * N + col0, acc0[r]);
    store_out(C + (size_t)row * N + col0 + 16, acc1[r]);
  }
}

// ---------------- Scores: attn[b,h,q,k] = mask ? -1e9 : (Q.K^T)/8 --------
__global__ void scores_kernel(const unsigned short* __restrict__ Qp,
                              const unsigned short* __restrict__ Kp,
                              const unsigned char* __restrict__ mask,
                              float* __restrict__ attn) {
  const int tid = threadIdx.x;
  const int w = tid >> 5, lane = tid & 31;
  const int laneLo = lane & 15, laneHi = lane >> 4;
  const unsigned int gt = blockIdx.x * 8u + w;     // tile id
  const int kt = gt & 127;
  const int qt = (gt >> 7) & 127;
  const int h  = (gt >> 14) & 15;
  const int b  = gt >> 18;

  const int q    = qt * 16 + laneLo;
  const int kcol = kt * 16 + laneLo;
  const size_t qoff = ((size_t)b * 2048 + q)    * 1024 + h * 64;
  const size_t koff = ((size_t)b * 2048 + kcol) * 1024 + h * 64;

  v8f acc = {};
#pragma unroll
  for (int dk = 0; dk < 64; dk += 32) {
    unsigned int afr[8];
    load_fragA(Qp + qoff + dk + laneHi * 16, afr);
    unsigned int bfr[8];
    load_b16(Kp + koff + dk + laneHi * 16, bfr);   // B: lane=N(k), half=K(d)
    acc = wmma_bf16(afr, bfr, acc);
  }

#pragma unroll
  for (int r = 0; r < 8; ++r) {
    int qi = qt * 16 + r + laneHi * 8;
    size_t moff = ((size_t)b * 2048 + qi) * 2048 + kcol;
    size_t ooff = (((size_t)b * 16 + h) * 2048 + qi) * 2048 + kcol;
    attn[ooff] = mask[moff] ? -1e9f : acc[r] * 0.125f;
  }
}

// ---------------- Row softmax in place (row = 2048) ----------------
__global__ void softmax_kernel(float* __restrict__ attn) {
  __shared__ float red[8];
  const size_t row = blockIdx.x;
  float* p = attn + row * 2048;
  const int tid = threadIdx.x;
  const int w = tid >> 5, lane = tid & 31;

  float v[8];
  float mx = -3.4e38f;
#pragma unroll
  for (int i = 0; i < 8; ++i) { v[i] = p[tid + i * 256]; mx = fmaxf(mx, v[i]); }
#pragma unroll
  for (int o = 16; o > 0; o >>= 1) mx = fmaxf(mx, __shfl_xor(mx, o, 32));
  if (lane == 0) red[w] = mx;
  __syncthreads();
  mx = red[0];
#pragma unroll
  for (int j = 1; j < 8; ++j) mx = fmaxf(mx, red[j]);
  __syncthreads();

  float s = 0.f;
#pragma unroll
  for (int i = 0; i < 8; ++i) { v[i] = __expf(v[i] - mx); s += v[i]; }
#pragma unroll
  for (int o = 16; o > 0; o >>= 1) s += __shfl_xor(s, o, 32);
  if (lane == 0) red[w] = s;
  __syncthreads();
  s = red[0];
#pragma unroll
  for (int j = 1; j < 8; ++j) s += red[j];
  const float inv = 1.f / s;
#pragma unroll
  for (int i = 0; i < 8; ++i) p[tid + i * 256] = v[i] * inv;
}

// ---------------- ctx[b,q,h*64+dv] = attn[b,h,q,:] . V[b,:,h,dv] ---------
// Block tile: 32(q) x 64(dv), waves 2x4. V tile staged transposed in LDS.
__global__ void ctx_kernel(const float* __restrict__ attn,
                           const unsigned short* __restrict__ Vp,
                           unsigned short* __restrict__ ctx) {
  __shared__ unsigned short lV[64 * 32];           // [dv][k]
  const int tid = threadIdx.x;
  const int w = tid >> 5, lane = tid & 31;
  const int laneLo = lane & 15, laneHi = lane >> 4;
  const int wave_m = w >> 2, wave_n = w & 3;
  const unsigned int bx = blockIdx.x;
  const int qt = bx & 63;
  const int h  = (bx >> 6) & 15;
  const int b  = bx >> 10;

  const int q = qt * 32 + wave_m * 16 + laneLo;
  const size_t arow = (((size_t)b * 16 + h) * 2048 + q) * 2048;
  const int dvl = tid & 63;                        // staging col
  const int kl0 = tid >> 6;                        // staging k row (0..3)

  v8f acc = {};
  for (int kk = 0; kk < 2048; kk += 32) {
#pragma unroll
    for (int p = 0; p < 8; ++p) {
      int k = kl0 + p * 4;
      lV[dvl * 32 + k] = Vp[((size_t)b * 2048 + kk + k) * 1024 + h * 64 + dvl];
    }
    __syncthreads();

    unsigned int afr[8];
    load_fragA(attn + arow + kk + laneHi * 16, afr);
    unsigned int bfr[8];
    const unsigned int* pb =
        reinterpret_cast<const unsigned int*>(&lV[(wave_n * 16 + laneLo) * 32 + laneHi * 16]);
#pragma unroll
    for (int j = 0; j < 8; ++j) bfr[j] = pb[j];

    acc = wmma_bf16(afr, bfr, acc);
    __syncthreads();
  }

  const int dv = wave_n * 16 + laneLo;
#pragma unroll
  for (int r = 0; r < 8; ++r) {
    int qi = qt * 32 + wave_m * 16 + r + laneHi * 8;
    ctx[((size_t)b * 2048 + qi) * 1024 + h * 64 + dv] = f32_to_bf16(acc[r]);
  }
}

// ---------------- residual + LayerNorm (row = 1024) ----------------
__global__ void layernorm_kernel(const float* __restrict__ fcout,
                                 const float* __restrict__ inQ,
                                 float* __restrict__ res) {
  __shared__ float red[8];
  const size_t row = blockIdx.x;
  const float* pf = fcout + row * 1024;
  const float* pq = inQ + row * 1024;
  const int tid = threadIdx.x;
  const int w = tid >> 5, lane = tid & 31;

  float x[4];
  float s = 0.f;
#pragma unroll
  for (int i = 0; i < 4; ++i) { x[i] = pf[tid + i * 256] + pq[tid + i * 256]; s += x[i]; }
#pragma unroll
  for (int o = 16; o > 0; o >>= 1) s += __shfl_xor(s, o, 32);
  if (lane == 0) red[w] = s;
  __syncthreads();
  s = red[0] + red[1] + red[2] + red[3] + red[4] + red[5] + red[6] + red[7];
  const float mu = s * (1.f / 1024.f);
  __syncthreads();

  float vs = 0.f;
#pragma unroll
  for (int i = 0; i < 4; ++i) { float d = x[i] - mu; vs += d * d; }
#pragma unroll
  for (int o = 16; o > 0; o >>= 1) vs += __shfl_xor(vs, o, 32);
  if (lane == 0) red[w] = vs;
  __syncthreads();
  vs = red[0] + red[1] + red[2] + red[3] + red[4] + red[5] + red[6] + red[7];
  const float inv = rsqrtf(vs * (1.f / 1024.f) + 1e-5f);
#pragma unroll
  for (int i = 0; i < 4; ++i)
    res[row * 1024 + tid + i * 256] = (x[i] - mu) * inv;
}

// ---------------- host launcher ----------------
extern "C" void kernel_launch(void* const* d_in, const int* in_sizes, int n_in,
                              void* d_out, int out_size, void* d_ws, size_t ws_size,
                              hipStream_t stream) {
  (void)in_sizes; (void)n_in; (void)out_size; (void)ws_size;
  const float*         inQ  = (const float*)d_in[0];
  const float*         inK  = (const float*)d_in[1];
  const float*         inV  = (const float*)d_in[2];
  const unsigned char* mask = (const unsigned char*)d_in[3];
  const float*         WQ   = (const float*)d_in[4];
  const float*         WK   = (const float*)d_in[5];
  const float*         WV   = (const float*)d_in[6];
  const float*         Wfc  = (const float*)d_in[7];

  float* out  = (float*)d_out;
  float* res  = out;                              // [4,2048,1024]
  float* attn = out + (size_t)8192 * 1024;        // [4,16,2048,2048]

  char* ws = (char*)d_ws;
  unsigned short* Qp   = (unsigned short*)(ws);                              // 16 MiB
  unsigned short* Kp   = (unsigned short*)(ws + (size_t)16 * 1024 * 1024);   // 16 MiB
  unsigned short* Vp   = (unsigned short*)(ws + (size_t)32 * 1024 * 1024);   // 16 MiB
  unsigned short* ctx  = (unsigned short*)(ws + (size_t)48 * 1024 * 1024);   // 16 MiB
  float*          fco  = (float*)(ws + (size_t)64 * 1024 * 1024);            // 32 MiB

  dim3 blk(256);
  // QKV projections: M=8192, N=1024, K=1024 -> (8192/64)*(1024/64)=2048 blocks
  gemm_bf16_wmma<float, unsigned short><<<2048, blk, 0, stream>>>(inQ, WQ, Qp, 8192, 1024, 1024);
  gemm_bf16_wmma<float, unsigned short><<<2048, blk, 0, stream>>>(inK, WK, Kp, 8192, 1024, 1024);
  gemm_bf16_wmma<float, unsigned short><<<2048, blk, 0, stream>>>(inV, WV, Vp, 8192, 1024, 1024);

  // scores: B*H*(2048/16)^2 = 1,048,576 wave-tiles / 8 per block
  scores_kernel<<<131072, blk, 0, stream>>>(Qp, Kp, mask, attn);
  // softmax: one block per (b,h,q) row
  softmax_kernel<<<131072, blk, 0, stream>>>(attn);
  // ctx: B*H*(2048/32) = 4096 blocks
  ctx_kernel<<<4096, blk, 0, stream>>>(attn, Vp, ctx);
  // output projection
  gemm_bf16_wmma<unsigned short, float><<<2048, blk, 0, stream>>>(ctx, Wfc, fco, 8192, 1024, 1024);
  // residual + LayerNorm
  layernorm_kernel<<<8192, blk, 0, stream>>>(fco, inQ, res);
}
